// aagmm_layer_21328807592695
// MI455X (gfx1250) — matrix operand order repaired and checked by reading, not compiled, not run on hardware
//
#include <hip/hip_runtime.h>
#include <stdint.h>

#define B_N 2048
#define K_N 128
#define D_N 512
#define KD  1024        // doubled reduction dim: [x^2 ; x] vs [w ; -2cw]
#define EPSV 1e-8f

typedef __attribute__((ext_vector_type(16))) __bf16 v16bf;
typedef __attribute__((ext_vector_type(8)))  float  v8f;

// ---- workspace layout (bytes) ----
static constexpr size_t OFF_A_HI = 0;                               // u16[B_N*KD] = 4 MiB
static constexpr size_t OFF_A_LO = OFF_A_HI + (size_t)B_N * KD * 2; // 4 MiB
static constexpr size_t OFF_B_HI = OFF_A_LO + (size_t)B_N * KD * 2; // u16[K_N*KD] = 256 KiB
static constexpr size_t OFF_B_LO = OFF_B_HI + (size_t)K_N * KD * 2;
static constexpr size_t OFF_Q    = OFF_B_LO + (size_t)K_N * KD * 2; // f32[128]
static constexpr size_t OFF_DSF  = OFF_Q    + 128 * 4;              // f32[128]
static constexpr size_t OFF_LDS  = OFF_DSF  + 128 * 4;              // f32[128]

// ---- bf16 split helpers (round-to-nearest-even) ----
__device__ __forceinline__ unsigned short f2bf_rne(float f) {
    unsigned int u = __float_as_uint(f);
    u += 0x7FFFu + ((u >> 16) & 1u);
    return (unsigned short)(u >> 16);
}
__device__ __forceinline__ float bf2f(unsigned short h) {
    return __uint_as_float(((unsigned int)h) << 16);
}
__device__ __forceinline__ void split_bf(float v, unsigned short& hi, unsigned short& lo) {
    hi = f2bf_rne(v);
    lo = f2bf_rne(v - bf2f(hi));
}

// ---------------------------------------------------------------------------
// Kernel 1: per-cluster weights -> B fragments (w , -2*c*w), plus q[k], det_sf[k]
// B-fragment layout for v_wmma_f32_16x16x32_bf16 (B is 32x16 K-major):
//   lane = (k%16) + ((r&8) ? 16 : 0), element pos = (r&7) + ((r&16) ? 8 : 0), r = dim within 32-chunk
// Stored as [ct][kc][lane][16] bf16.
// ---------------------------------------------------------------------------
__global__ void prep_w_kernel(const float* __restrict__ centers,
                              const float* __restrict__ Dm,
                              uint16_t* __restrict__ Bhi, uint16_t* __restrict__ Blo,
                              float* __restrict__ qout, float* __restrict__ detsf) {
    int k = blockIdx.x;            // cluster
    int tid = threadIdx.x;         // 256 threads
    int ct = k >> 4, laneN = k & 15;
    float logsum = 0.f, qsum = 0.f;
    for (int d = tid; d < D_N; d += 256) {
        float Dab = fabsf(Dm[(size_t)k * D_N + d]) + EPSV;
        float w   = 1.0f / Dab;
        float c   = centers[(size_t)k * D_N + d];
        logsum += logf(Dab);
        qsum   += c * c * w;
        float vals[2] = { w, -2.0f * c * w };
        int   js[2]   = { d, D_N + d };
#pragma unroll
        for (int t = 0; t < 2; ++t) {
            int j = js[t];
            int kc = j >> 5, r = j & 31;
            int lane = laneN + ((r & 8) ? 16 : 0);
            int pos  = (r & 7) + ((r & 16) ? 8 : 0);
            size_t off = (((size_t)(ct * 32 + kc)) * 32 + lane) * 16 + pos;
            unsigned short hi, lo;
            split_bf(vals[t], hi, lo);
            Bhi[off] = hi; Blo[off] = lo;
        }
    }
    __shared__ float s1[256], s2[256];
    s1[tid] = logsum; s2[tid] = qsum;
    __syncthreads();
    for (int o = 128; o > 0; o >>= 1) {
        if (tid < o) { s1[tid] += s1[tid + o]; s2[tid] += s2[tid + o]; }
        __syncthreads();
    }
    if (tid == 0) { detsf[k] = -0.5f * s1[0]; qout[k] = s2[0]; }
}

// ---------------------------------------------------------------------------
// Kernel 2: log det_scale = det_sf - max_k(det_sf)
// ---------------------------------------------------------------------------
__global__ void prep_scale_kernel(const float* __restrict__ detsf,
                                  float* __restrict__ ldsc) {
    int t = threadIdx.x;           // 128 threads
    __shared__ float sm[128];
    float v = detsf[t];
    sm[t] = v;
    __syncthreads();
    for (int o = 64; o > 0; o >>= 1) {
        if (t < o) sm[t] = fmaxf(sm[t], sm[t + o]);
        __syncthreads();
    }
    ldsc[t] = v - sm[0];
}

// ---------------------------------------------------------------------------
// Kernel 3: x -> A fragments [x^2 ; x] hi/lo, in the 16-bit A 16x32 lane layout:
//   r<8: lane=m,pos=r | 8<=r<16: lane=m+16,pos=r-8 | 16<=r<24: lane=m,pos=r-8
//   | r>=24: lane=m+16,pos=r-16   (m = row%16)
// Stored as [rt][kc][lane][16] bf16.
// ---------------------------------------------------------------------------
__global__ void prep_x_kernel(const float* __restrict__ x,
                              uint16_t* __restrict__ Ahi, uint16_t* __restrict__ Alo) {
    int b = blockIdx.x, tid = threadIdx.x;   // one row per block
    int rt = b >> 4, m = b & 15;
    for (int j = tid; j < KD; j += 256) {
        float v;
        if (j < D_N) { float xv = x[(size_t)b * D_N + j]; v = xv * xv; }
        else         { v = x[(size_t)b * D_N + (j - D_N)]; }
        int kc = j >> 5, r = j & 31;
        int lane, pos;
        if      (r < 8)  { lane = m;      pos = r;      }
        else if (r < 16) { lane = m + 16; pos = r - 8;  }
        else if (r < 24) { lane = m;      pos = r - 8;  }
        else             { lane = m + 16; pos = r - 16; }
        size_t off = (((size_t)(rt * 32 + kc)) * 32 + lane) * 16 + pos;
        unsigned short hi, lo;
        split_bf(v, hi, lo);
        Ahi[off] = hi; Alo[off] = lo;
    }
}

// ---------------------------------------------------------------------------
// Kernel 4: WMMA GEMM (split-bf16, 3 matmuls ~= f32) + fused log-softmax epilogue.
// One wave per 16-row tile; 8 x v8f accumulators hold all 128 clusters.
// ---------------------------------------------------------------------------
__global__ void gemm_softmax_kernel(const uint16_t* __restrict__ Ahi,
                                    const uint16_t* __restrict__ Alo,
                                    const uint16_t* __restrict__ Bhi,
                                    const uint16_t* __restrict__ Blo,
                                    const float* __restrict__ q,
                                    const float* __restrict__ ldsc,
                                    float* __restrict__ resp,
                                    float* __restrict__ cdist) {
    const int lane = threadIdx.x & 31;
    const int tile = blockIdx.x * 8 + (threadIdx.x >> 5);   // 16 blocks x 8 waves = 128 tiles

    v8f acc[8] = {};
    const v16bf* Ah = (const v16bf*)Ahi;
    const v16bf* Al = (const v16bf*)Alo;
    const v16bf* Bh = (const v16bf*)Bhi;
    const v16bf* Bl = (const v16bf*)Blo;

    for (int kc = 0; kc < 32; ++kc) {
        v16bf ah = Ah[((size_t)tile * 32 + kc) * 32 + lane];
        v16bf al = Al[((size_t)tile * 32 + kc) * 32 + lane];
#pragma unroll
        for (int ct = 0; ct < 8; ++ct) {
            v16bf bh = Bh[((size_t)ct * 32 + kc) * 32 + lane];
            v16bf bl = Bl[((size_t)ct * 32 + kc) * 32 + lane];
            acc[ct] = __builtin_amdgcn_wmma_f32_16x16x32_bf16(false, ah, false, bh,
                                                              (short)0, acc[ct], false, false);
            acc[ct] = __builtin_amdgcn_wmma_f32_16x16x32_bf16(false, ah, false, bl,
                                                              (short)0, acc[ct], false, false);
            acc[ct] = __builtin_amdgcn_wmma_f32_16x16x32_bf16(false, al, false, bh,
                                                              (short)0, acc[ct], false, false);
        }
    }

    // Per-lane cluster constants: k = ct*16 + lane%16
    float qv[8], ldv[8];
#pragma unroll
    for (int ct = 0; ct < 8; ++ct) {
        int kk = ct * 16 + (lane & 15);
        qv[ct]  = q[kk];
        ldv[ct] = ldsc[kk];
    }

    // C layout: acc[ct][i] -> row = tile*16 + i + (lane<16?0:8), col = ct*16 + lane%16
#pragma unroll
    for (int i = 0; i < 8; ++i) {
        float s[8];
        float bs = -3.4e38f; int bk = 0; float bd = 0.f;
#pragma unroll
        for (int ct = 0; ct < 8; ++ct) {
            float dist = acc[ct][i] + qv[ct];        // dist_sq
            float sv   = ldv[ct] - 0.5f * dist;      // log det_scale + expo
            s[ct] = sv;
            if (sv > bs) { bs = sv; bk = ct * 16 + (lane & 15); bd = dist; }
        }
        // reduce (max-s, argmin-k tiebreak, carry dist) across the 16-lane group
#pragma unroll
        for (int msk = 8; msk >= 1; msk >>= 1) {
            float os = __shfl_xor(bs, msk, 32);
            int   ok = __shfl_xor(bk, msk, 32);
            float od = __shfl_xor(bd, msk, 32);
            if (os > bs || (os == bs && ok < bk)) { bs = os; bk = ok; bd = od; }
        }
        float M = bs;
        float Z = 0.f;
#pragma unroll
        for (int ct = 0; ct < 8; ++ct) Z += expf(s[ct] - M);
#pragma unroll
        for (int msk = 8; msk >= 1; msk >>= 1) Z += __shfl_xor(Z, msk, 32);
        float lZ = logf(Z);

        int m   = i + ((lane & 16) ? 8 : 0);
        int row = tile * 16 + m;
#pragma unroll
        for (int ct = 0; ct < 8; ++ct)
            resp[(size_t)row * K_N + ct * 16 + (lane & 15)] =
                fmaxf(s[ct] - M - lZ, -18.4206807f);   // log(clip(resp, 1e-8))
        if ((lane & 15) == 0)
            cdist[row] = sqrtf(fmaxf(bd, 0.f));
    }
}

extern "C" void kernel_launch(void* const* d_in, const int* in_sizes, int n_in,
                              void* d_out, int out_size, void* d_ws, size_t ws_size,
                              hipStream_t stream) {
    (void)in_sizes; (void)n_in; (void)out_size; (void)ws_size;
    const float* x       = (const float*)d_in[0];
    const float* centers = (const float*)d_in[1];
    const float* Dm      = (const float*)d_in[2];

    char* ws = (char*)d_ws;
    uint16_t* Ahi  = (uint16_t*)(ws + OFF_A_HI);
    uint16_t* Alo  = (uint16_t*)(ws + OFF_A_LO);
    uint16_t* Bhi  = (uint16_t*)(ws + OFF_B_HI);
    uint16_t* Blo  = (uint16_t*)(ws + OFF_B_LO);
    float*    qbuf = (float*)(ws + OFF_Q);
    float*    dsf  = (float*)(ws + OFF_DSF);
    float*    ldsb = (float*)(ws + OFF_LDS);

    float* resp  = (float*)d_out;                 // [2048,128]
    float* cdist = resp + (size_t)B_N * K_N;      // [2048]

    prep_w_kernel<<<K_N, 256, 0, stream>>>(centers, Dm, Bhi, Blo, qbuf, dsf);
    prep_scale_kernel<<<1, 128, 0, stream>>>(dsf, ldsb);
    prep_x_kernel<<<B_N, 256, 0, stream>>>(x, Ahi, Alo);
    gemm_softmax_kernel<<<16, 256, 0, stream>>>(Ahi, Alo, Bhi, Blo, qbuf, ldsb, resp, cdist);
}
